// Transformer_45990509805786
// MI455X (gfx1250) — matrix operand order repaired and checked by reading, not compiled, not run on hardware
//
#include <hip/hip_runtime.h>
#include <hip/hip_bf16.h>
#include <math.h>

typedef __attribute__((ext_vector_type(16))) _Float16 v16h;
typedef __attribute__((ext_vector_type(8)))  float    v8f;

#define Bb    8
#define Ss    1024
#define Dd    512
#define Hh    8
#define Vv    16384
#define MLPd  1024
#define NMIDn 4
#define Mrows (Bb*Ss)   /* 8192 */

// ---------------------------------------------------------------------------
// CDNA5 async global->LDS copy (16B per lane, tracked by ASYNCcnt).
// ldsOff = low 32 bits of the generic LDS address (= LDS byte offset).
// ---------------------------------------------------------------------------
__device__ __forceinline__ void async_cp_b128(unsigned ldsOff, const _Float16* gsrc) {
  asm volatile("global_load_async_to_lds_b128 %0, %1, off"
               :: "v"(ldsOff), "v"(gsrc) : "memory");
}
__device__ __forceinline__ void async_wait0() {
  asm volatile("s_wait_asynccnt 0x0" ::: "memory");
}

// ---------------------------------------------------------------------------
// WMMA fragment loaders from LDS tiles (CDNA5 16x16x32 f16 layouts).
// A (16xK=32): lane 0-15 = M, half-wave splits K at 8; VGPR j holds K pair
//   kp = (j>=4?16:0) + half*8 + (j&3)*2
// B (K=32x16): lane 0-15 = N, half-wave splits K at 8; VGPR j holds K pair
//   kp = (j>=4?16:0) + half*8 + (j&3)*2
// stride in _Float16 units, must be even.
// ---------------------------------------------------------------------------
__device__ __forceinline__ v16h load_frag_a(const _Float16* base, int stride, int lane) {
  int m = lane & 15, half = lane >> 4;
  const _Float16* rowp = base + m * stride;
  union { v16h v; unsigned int u[8]; } r;
#pragma unroll
  for (int j = 0; j < 8; ++j) {
    int kp = ((j & 4) ? 16 : 0) + half * 8 + (j & 3) * 2;
    r.u[j] = *(const unsigned int*)(rowp + kp);
  }
  return r.v;
}

__device__ __forceinline__ v16h load_frag_b(const _Float16* base, int stride, int lane) {
  int n = lane & 15, half = lane >> 4;
  const _Float16* rowp = base + n * stride;
  union { v16h v; unsigned int u[8]; } r;
#pragma unroll
  for (int j = 0; j < 8; ++j) {
    int kp = ((j & 4) ? 16 : 0) + half * 8 + (j & 3) * 2;
    r.u[j] = *(const unsigned int*)(rowp + kp);
  }
  return r.v;
}

#define WMMA_F16(A, Bv, C) \
  __builtin_amdgcn_wmma_f32_16x16x32_f16(false, (A), false, (Bv), (short)0, (C), false, false)

// ---------------------------------------------------------------------------
// Weight pre-conversion f32 -> f16 (runs once; keeps weights L2-resident at
// half the bytes).
// ---------------------------------------------------------------------------
__global__ __launch_bounds__(256) void f32_to_f16_kernel(
    const float* __restrict__ src, _Float16* __restrict__ dst, long long n) {
  long long i = (long long)blockIdx.x * 256 + threadIdx.x;
  long long st = (long long)gridDim.x * 256;
  for (; i < n; i += st) dst[i] = (_Float16)src[i];
}

// ---------------------------------------------------------------------------
// Embedding + sinusoidal PE:  e[b,s,:] = emb[x[b,s],:] + pe(s,:)
// ---------------------------------------------------------------------------
__global__ __launch_bounds__(256) void embed_pe_kernel(
    const int* __restrict__ x, const float* __restrict__ emb, float* __restrict__ e) {
  int row = blockIdx.x;                 // b*S + s
  int pos = row & (Ss - 1);
  int tok = x[row];
  const float* er = emb + (size_t)tok * Dd;
  float* outr = e + (size_t)row * Dd;
  for (int d = threadIdx.x; d < Dd; d += 256) {
    float di  = (float)(d & ~1);
    float div = __expf(di * (-9.210340371976184f / (float)Dd)); // ln(10000)
    float ang = (float)pos * div;
    float pe  = (d & 1) ? __cosf(ang) : __sinf(ang);
    outr[d] = er[d] + pe;
  }
}

// ---------------------------------------------------------------------------
// LayerNorm (population var, eps=1e-5), optional residual, optional ReLU,
// f16 output. One block per row of D=512, 256 threads (2 elems each).
// ---------------------------------------------------------------------------
__global__ __launch_bounds__(256) void ln_kernel(
    const float* __restrict__ xin, const float* __restrict__ res,
    const float* __restrict__ g, const float* __restrict__ bta,
    _Float16* __restrict__ outh, int relu) {
  __shared__ float s1[8], s2[8];
  int row = blockIdx.x;
  const float* xr = xin + (size_t)row * Dd;
  const float* rr = res ? res + (size_t)row * Dd : nullptr;
  int t = threadIdx.x;
  float v0 = xr[t]       + (rr ? rr[t]       : 0.f);
  float v1 = xr[t + 256] + (rr ? rr[t + 256] : 0.f);
  float ps = v0 + v1, pq = v0 * v0 + v1 * v1;
#pragma unroll
  for (int m = 16; m >= 1; m >>= 1) {
    ps += __shfl_xor(ps, m, 32);
    pq += __shfl_xor(pq, m, 32);
  }
  int wave = t >> 5;
  if ((t & 31) == 0) { s1[wave] = ps; s2[wave] = pq; }
  __syncthreads();
  float sum = 0.f, sq = 0.f;
#pragma unroll
  for (int i = 0; i < 8; ++i) { sum += s1[i]; sq += s2[i]; }
  float mean = sum * (1.f / Dd);
  float var  = sq  * (1.f / Dd) - mean * mean;
  float rstd = rsqrtf(var + 1e-5f);
  float o0 = (v0 - mean) * rstd * g[t]       + bta[t];
  float o1 = (v1 - mean) * rstd * g[t + 256] + bta[t + 256];
  if (relu) { o0 = fmaxf(o0, 0.f); o1 = fmaxf(o1, 0.f); }
  _Float16* outr = outh + (size_t)row * Dd;
  outr[t] = (_Float16)o0;
  outr[t + 256] = (_Float16)o1;
}

// ---------------------------------------------------------------------------
// WMMA GEMM:  C[M,N] = A_f16[M,K] @ W_f16[N,K]^T (+bias)(ReLU)(+res)
// Block tile 128x128x32, 8 waves (4x2), wave tile 32x64 (2x4 WMMA accs).
// Double-buffered LDS fed by GLOBAL_LOAD_ASYNC_TO_LDS_B128 (ASYNCcnt).
// All 6 fragments loaded into distinct registers before the 8 WMMAs so the
// ds_loads issue as one group and the WMMAs fire back-to-back.
// ---------------------------------------------------------------------------
__global__ __launch_bounds__(256) void wmma_gemm_kernel(
    const _Float16* __restrict__ A, const _Float16* __restrict__ W,
    const float* __restrict__ bias,
    float* __restrict__ outF32, _Float16* __restrict__ outF16,
    const float* __restrict__ resF32,
    int M, int N, int K, int relu,
    size_t strideW, size_t strideB, size_t strideO) {
  __shared__ __align__(16) _Float16 As[2][128][40];
  __shared__ __align__(16) _Float16 Bs[2][128][40];
  int tid = threadIdx.x, lane = tid & 31, wave = tid >> 5;
  int wm = wave >> 1, wn = wave & 1;
  int blockM = blockIdx.y * 128, blockN = blockIdx.x * 128;
  int z = blockIdx.z;
  const _Float16* Wp = W + (size_t)z * strideW;
  const float* bp = bias ? bias + (size_t)z * strideB : nullptr;

  // each thread copies 4 x 16B per tile: A rows rs, rs+64; W rows rs, rs+64
  int rs = tid >> 2, cs = (tid & 3) * 8;
  const _Float16* aRow0 = A  + (size_t)(blockM + rs) * K + cs;
  const _Float16* aRow1 = aRow0 + (size_t)64 * K;
  const _Float16* bRow0 = Wp + (size_t)(blockN + rs) * K + cs;
  const _Float16* bRow1 = bRow0 + (size_t)64 * K;

  auto stage = [&](int buf, int k0) {
    async_cp_b128((unsigned)(size_t)&As[buf][rs][cs],      aRow0 + k0);
    async_cp_b128((unsigned)(size_t)&As[buf][rs + 64][cs], aRow1 + k0);
    async_cp_b128((unsigned)(size_t)&Bs[buf][rs][cs],      bRow0 + k0);
    async_cp_b128((unsigned)(size_t)&Bs[buf][rs + 64][cs], bRow1 + k0);
  };

  v8f acc[2][4] = {};
  int nk = K >> 5;
  stage(0, 0);
  for (int kt = 0; kt < nk; ++kt) {
    int buf = kt & 1;
    async_wait0();        // this wave's copies for tile kt are in LDS
    __syncthreads();      // everyone's copies done; prev buffer free
    if (kt + 1 < nk) stage(buf ^ 1, (kt + 1) * 32);
    // load ALL fragments first (distinct registers), then 8 WMMAs
    v16h a0 = load_frag_a(&As[buf][wm * 32][0],      40, lane);
    v16h a1 = load_frag_a(&As[buf][wm * 32 + 16][0], 40, lane);
    v16h bf0 = load_frag_b(&Bs[buf][wn * 64][0],      40, lane);
    v16h bf1 = load_frag_b(&Bs[buf][wn * 64 + 16][0], 40, lane);
    v16h bf2 = load_frag_b(&Bs[buf][wn * 64 + 32][0], 40, lane);
    v16h bf3 = load_frag_b(&Bs[buf][wn * 64 + 48][0], 40, lane);
    acc[0][0] = WMMA_F16(a0, bf0, acc[0][0]);
    acc[1][0] = WMMA_F16(a1, bf0, acc[1][0]);
    acc[0][1] = WMMA_F16(a0, bf1, acc[0][1]);
    acc[1][1] = WMMA_F16(a1, bf1, acc[1][1]);
    acc[0][2] = WMMA_F16(a0, bf2, acc[0][2]);
    acc[1][2] = WMMA_F16(a1, bf2, acc[1][2]);
    acc[0][3] = WMMA_F16(a0, bf3, acc[0][3]);
    acc[1][3] = WMMA_F16(a1, bf3, acc[1][3]);
  }

  int n16 = lane & 15, half = lane >> 4;
  float*    o32 = outF32 ? outF32 + (size_t)z * strideO : nullptr;
  _Float16* o16 = outF16 ? outF16 + (size_t)z * strideO : nullptr;
#pragma unroll
  for (int i = 0; i < 2; ++i)
#pragma unroll
    for (int j = 0; j < 4; ++j) {
      int col = blockN + wn * 64 + j * 16 + n16;
      float bv = bp ? bp[col] : 0.f;
#pragma unroll
      for (int r = 0; r < 8; ++r) {
        int row = blockM + wm * 32 + i * 16 + half * 8 + r;
        size_t idx = (size_t)row * N + col;
        float vvv = acc[i][j][r] + bv;
        if (relu)   vvv = fmaxf(vvv, 0.f);
        if (resF32) vvv += resF32[idx];
        if (o32) o32[idx] = vvv;
        if (o16) o16[idx] = (_Float16)vvv;
      }
    }
}

// ---------------------------------------------------------------------------
// Flash attention per (b,h,16-query tile). 8 waves: every wave redundantly
// computes the 16x32 score block (WMMA over D), online softmax + entropy
// stats; each wave owns a 64-wide e-slice of the w@V accumulation (WMMA).
// Mask (faithful to source): t <= s  OR  pad(key t)  -> -1e9.
// Heads are SUMMED into attnOut via f32 atomics.
// Q and K tiles staged with async global->LDS copies.
// ---------------------------------------------------------------------------
__global__ __launch_bounds__(256) void flash_attn_kernel(
    const _Float16* __restrict__ Qh, const _Float16* __restrict__ Kh,
    const _Float16* __restrict__ Vh, const int* __restrict__ x,
    float* __restrict__ attnOut, float* __restrict__ entAcc) {
  __shared__ __align__(16) _Float16 Qs[16][520];   // 16 queries x D
  __shared__ __align__(16) _Float16 Ks[32][72];    // 32 keys x 64-dk chunk
  __shared__ __align__(16) _Float16 Vt[512][34];   // transposed: [e][key]
  __shared__ __align__(16) _Float16 Ps[8][16][34]; // per-wave p tile
  __shared__ int padf[32];

  int tid = threadIdx.x, lane = tid & 31, wave = tid >> 5;
  int n16 = lane & 15, half = lane >> 4;
  int bh = blockIdx.y;
  int b = bh & (Bb - 1);
  int h = bh >> 3;
  int qBase = blockIdx.x * 16;
  size_t headOff = (size_t)h * Mrows * Dd;
  size_t rowOff  = (size_t)b * Ss;

  // stage Q tile (16 x 512) via async copies
#pragma unroll
  for (int it = 0; it < 4; ++it) {
    int i = tid + it * 256;
    int r = i >> 6, c = (i & 63) * 8;
    async_cp_b128((unsigned)(size_t)&Qs[r][c],
                  Qh + headOff + (rowOff + qBase + r) * (size_t)Dd + c);
  }
  async_wait0();

  v8f oacc[4] = {};
  float rm[8], rl[8], rt[8];
#pragma unroll
  for (int r = 0; r < 8; ++r) { rm[r] = -1e30f; rl[r] = 0.f; rt[r] = 0.f; }
  const float scl = 0.044194173824159216f;  // 1/sqrt(512)

  for (int kc = 0; kc < Ss / 32; ++kc) {
    int keyBase = kc * 32;
    __syncthreads();  // previous chunk's Vt reads finished before restage
    if (tid < 32) padf[tid] = (x[rowOff + keyBase + tid] == 0) ? 1 : 0;
    // stage V chunk transposed: Vt[e][key] (manual: needs scatter)
#pragma unroll
    for (int it = 0; it < 8; ++it) {
      int i = tid + it * 256;
      int key = i >> 6, c = (i & 63) * 8;
      union { uint4 q; _Float16 hv[8]; } u;
      u.q = *(const uint4*)(Vh + headOff + (rowOff + keyBase + key) * (size_t)Dd + c);
#pragma unroll
      for (int j2 = 0; j2 < 8; ++j2) Vt[c + j2][key] = u.hv[j2];
    }

    v8f s0 = {}, s1 = {};
    for (int dk0 = 0; dk0 < Dd; dk0 += 64) {
      __syncthreads();
      {
        int key = tid >> 3, c = (tid & 7) * 8;
        async_cp_b128((unsigned)(size_t)&Ks[key][c],
                      Kh + headOff + (rowOff + keyBase + key) * (size_t)Dd + dk0 + c);
      }
      async_wait0();
      __syncthreads();
      // load all 6 fragments for the two K-steps, then 4 WMMAs
      v16h aqA  = load_frag_a(&Qs[0][dk0],      520, lane);
      v16h aqB  = load_frag_a(&Qs[0][dk0 + 32], 520, lane);
      v16h kb0A = load_frag_b(&Ks[0][0],   72, lane);
      v16h kb1A = load_frag_b(&Ks[16][0],  72, lane);
      v16h kb0B = load_frag_b(&Ks[0][32],  72, lane);
      v16h kb1B = load_frag_b(&Ks[16][32], 72, lane);
      s0 = WMMA_F16(aqA, kb0A, s0);
      s1 = WMMA_F16(aqA, kb1A, s1);
      s0 = WMMA_F16(aqB, kb0B, s0);
      s1 = WMMA_F16(aqB, kb1B, s1);
    }

    // online softmax update + entropy stats; build p tile
#pragma unroll
    for (int r = 0; r < 8; ++r) {
      int qrow = qBase + half * 8 + r;
      int t0 = keyBase + n16, t1 = keyBase + 16 + n16;
      float a0 = s0[r] * scl, a1 = s1[r] * scl;
      if (t0 <= qrow || padf[n16])      a0 = -1e9f;
      if (t1 <= qrow || padf[16 + n16]) a1 = -1e9f;
      float mx = fmaxf(a0, a1);
#pragma unroll
      for (int m = 8; m >= 1; m >>= 1) mx = fmaxf(mx, __shfl_xor(mx, m, 32));
      float m_new = fmaxf(rm[r], mx);
      float so = __expf(rm[r] - m_new);
      float p0 = __expf(a0 - m_new), p1 = __expf(a1 - m_new);
      float psum = p0 + p1, pws = p0 * a0 + p1 * a1;
#pragma unroll
      for (int m = 8; m >= 1; m >>= 1) {
        psum += __shfl_xor(psum, m, 32);
        pws  += __shfl_xor(pws,  m, 32);
      }
      rl[r] = rl[r] * so + psum;
      rt[r] = rt[r] * so + pws;
      rm[r] = m_new;
#pragma unroll
      for (int j = 0; j < 4; ++j) oacc[j][r] *= so;
      Ps[wave][half * 8 + r][n16]      = (_Float16)p0;
      Ps[wave][half * 8 + r][16 + n16] = (_Float16)p1;
    }

    // accumulate w @ V for this wave's e-slice (all fragments first)
    v16h pa  = load_frag_a(&Ps[wave][0][0], 34, lane);
    v16h vb0 = load_frag_b(&Vt[wave * 64][0],      34, lane);
    v16h vb1 = load_frag_b(&Vt[wave * 64 + 16][0], 34, lane);
    v16h vb2 = load_frag_b(&Vt[wave * 64 + 32][0], 34, lane);
    v16h vb3 = load_frag_b(&Vt[wave * 64 + 48][0], 34, lane);
    oacc[0] = WMMA_F16(pa, vb0, oacc[0]);
    oacc[1] = WMMA_F16(pa, vb1, oacc[1]);
    oacc[2] = WMMA_F16(pa, vb2, oacc[2]);
    oacc[3] = WMMA_F16(pa, vb3, oacc[3]);
  }

  // finalize: divide by Z, sum heads via atomics, entropy via logsumexp id
#pragma unroll
  for (int r = 0; r < 8; ++r) {
    float inv = 1.f / rl[r];
    if (wave == 0 && n16 == 0) {
      float ent = -(rt[r] * inv) + rm[r] + __logf(rl[r]);
      atomicAdd(entAcc, ent);
    }
    size_t orow = (rowOff + qBase + half * 8 + r) * (size_t)Dd;
#pragma unroll
    for (int j = 0; j < 4; ++j)
      atomicAdd(&attnOut[orow + wave * 64 + j * 16 + n16], oacc[j][r] * inv);
  }
}

__global__ void zero_f32_kernel(float* p, long long n) {
  long long i = (long long)blockIdx.x * blockDim.x + threadIdx.x;
  long long st = (long long)gridDim.x * blockDim.x;
  for (; i < n; i += st) p[i] = 0.f;
}

__global__ void ent_finalize_kernel(const float* __restrict__ entAcc,
                                    float* __restrict__ out) {
  out[0] = entAcc[0] * (1.f / (float)(Bb * Hh * Ss));
}

// ---------------------------------------------------------------------------
extern "C" void kernel_launch(void* const* d_in, const int* in_sizes, int n_in,
                              void* d_out, int out_size, void* d_ws, size_t ws_size,
                              hipStream_t stream) {
  (void)in_sizes; (void)n_in; (void)out_size; (void)ws_size;
  const int*   x    = (const int*)d_in[0];
  const float* emb  = (const float*)d_in[1];
  const float* Wq   = (const float*)d_in[2];
  const float* bq   = (const float*)d_in[3];
  const float* Wk   = (const float*)d_in[4];
  const float* bk   = (const float*)d_in[5];
  const float* Wv   = (const float*)d_in[6];
  const float* bv   = (const float*)d_in[7];
  const float* ln_g = (const float*)d_in[8];
  const float* ln_b = (const float*)d_in[9];
  const float* W0   = (const float*)d_in[10];
  const float* b0   = (const float*)d_in[11];
  const float* Wmid = (const float*)d_in[12];
  const float* bmid = (const float*)d_in[13];
  const float* Wout = (const float*)d_in[14];
  const float* bout = (const float*)d_in[15];
  float* logits = (float*)d_out;

  char* ws = (char*)d_ws;
  size_t off = 0;
  auto alloc = [&](size_t bytes) -> char* {
    char* p = ws + off;
    off += (bytes + 255) & ~(size_t)255;
    return p;
  };
  float*    e32    = (float*)   alloc((size_t)Mrows * Dd * 4);
  _Float16* h16    = (_Float16*)alloc((size_t)Mrows * Dd * 2);
  _Float16* Q16    = (_Float16*)alloc((size_t)Hh * Mrows * Dd * 2);
  _Float16* K16    = (_Float16*)alloc((size_t)Hh * Mrows * Dd * 2);
  _Float16* V16    = (_Float16*)alloc((size_t)Hh * Mrows * Dd * 2);
  float*    attn   = (float*)   alloc((size_t)Mrows * Dd * 4);
  float*    y32    = (float*)   alloc((size_t)Mrows * MLPd * 4);
  _Float16* yA     = (_Float16*)alloc((size_t)Mrows * MLPd * 2);
  _Float16* yB     = (_Float16*)alloc((size_t)Mrows * MLPd * 2);
  _Float16* Wq16   = (_Float16*)alloc((size_t)Hh * Dd * Dd * 2);
  _Float16* Wk16   = (_Float16*)alloc((size_t)Hh * Dd * Dd * 2);
  _Float16* Wv16   = (_Float16*)alloc((size_t)Hh * Dd * Dd * 2);
  _Float16* W016   = (_Float16*)alloc((size_t)MLPd * Dd * 2);
  _Float16* Wmid16 = (_Float16*)alloc((size_t)NMIDn * MLPd * MLPd * 2);
  _Float16* Wout16 = (_Float16*)alloc((size_t)Vv * MLPd * 2);
  float*    entA   = (float*)   alloc(256);

  // 0) one-time weight conversion to f16 (stays L2-resident at ~53MB)
  f32_to_f16_kernel<<<4096, 256, 0, stream>>>(Wq,   Wq16,   (long long)Hh * Dd * Dd);
  f32_to_f16_kernel<<<4096, 256, 0, stream>>>(Wk,   Wk16,   (long long)Hh * Dd * Dd);
  f32_to_f16_kernel<<<4096, 256, 0, stream>>>(Wv,   Wv16,   (long long)Hh * Dd * Dd);
  f32_to_f16_kernel<<<2048, 256, 0, stream>>>(W0,   W016,   (long long)MLPd * Dd);
  f32_to_f16_kernel<<<4096, 256, 0, stream>>>(Wmid, Wmid16, (long long)NMIDn * MLPd * MLPd);
  f32_to_f16_kernel<<<8192, 256, 0, stream>>>(Wout, Wout16, (long long)Vv * MLPd);

  // 1) embedding + PE, 2) LN -> h (f16)
  embed_pe_kernel<<<Mrows, 256, 0, stream>>>(x, emb, e32);
  ln_kernel<<<Mrows, 256, 0, stream>>>(e32, nullptr, ln_g, ln_b, h16, 0);

  // 3) QKV projections, batched across H via blockIdx.z
  dim3 gq(Dd / 128, Mrows / 128, Hh);
  wmma_gemm_kernel<<<gq, 256, 0, stream>>>(h16, Wq16, bq, nullptr, Q16, nullptr,
      Mrows, Dd, Dd, 0, (size_t)Dd * Dd, (size_t)Dd, (size_t)Mrows * Dd);
  wmma_gemm_kernel<<<gq, 256, 0, stream>>>(h16, Wk16, bk, nullptr, K16, nullptr,
      Mrows, Dd, Dd, 0, (size_t)Dd * Dd, (size_t)Dd, (size_t)Mrows * Dd);
  wmma_gemm_kernel<<<gq, 256, 0, stream>>>(h16, Wv16, bv, nullptr, V16, nullptr,
      Mrows, Dd, Dd, 0, (size_t)Dd * Dd, (size_t)Dd, (size_t)Mrows * Dd);

  // 4) flash attention (heads summed into attn via atomics)
  zero_f32_kernel<<<2048, 256, 0, stream>>>(attn, (long long)Mrows * Dd);
  zero_f32_kernel<<<1, 256, 0, stream>>>(entA, 64);
  dim3 ga(Ss / 16, Bb * Hh);
  flash_attn_kernel<<<ga, 256, 0, stream>>>(Q16, K16, V16, x, attn, entA);

  // 5) y = relu(LN(attn + e)) -> f16 (reuse h16)
  ln_kernel<<<Mrows, 256, 0, stream>>>(attn, e32, ln_g, ln_b, h16, 1);

  // 6) y = y @ W0^T + b0   (f32 + f16 copies)
  dim3 g0(MLPd / 128, Mrows / 128, 1);
  wmma_gemm_kernel<<<g0, 256, 0, stream>>>(h16, W016, b0, y32, yA, nullptr,
      Mrows, MLPd, Dd, 0, 0, 0, 0);

  // 7) y = y + relu(y @ Wmid[i]^T + bmid[i])  (ping-pong f16 buffers)
  for (int i = 0; i < NMIDn; ++i) {
    _Float16* src = (i % 2 == 0) ? yA : yB;
    _Float16* dst = (i % 2 == 0) ? yB : yA;
    wmma_gemm_kernel<<<g0, 256, 0, stream>>>(src,
        Wmid16 + (size_t)i * MLPd * MLPd, bmid + (size_t)i * MLPd,
        y32, dst, y32, Mrows, MLPd, MLPd, 1, 0, 0, 0);
  }

  // 8) logits = y @ Wout^T + bout -> d_out  (NMID=4 -> final f16 is in yA)
  dim3 gl(Vv / 128, Mrows / 128, 1);
  wmma_gemm_kernel<<<gl, 256, 0, stream>>>(yA, Wout16, bout, logits, nullptr,
      nullptr, Mrows, Vv, MLPd, 0, 0, 0, 0);

  // 9) entropy mean -> last output element
  ent_finalize_kernel<<<1, 1, 0, stream>>>(entA, logits + (size_t)Mrows * Vv);
}
